// NATLayer_79688823210199
// MI455X (gfx1250) — compile-verified, matrix operand order checked by reading, not run on hardware
//
#include <hip/hip_runtime.h>
#include <hip/hip_bf16.h>

// Problem constants (from reference)
#define BATCH   8
#define NSEQ    16384              // H*W = 128*128
#define CDIM    128
#define HEADS   4
#define DHEAD   32                 // C / HEADS
#define KS      7
#define MTOT    (BATCH * NSEQ)     // 131072 rows

typedef __bf16 bf16;
typedef __attribute__((ext_vector_type(16))) __bf16 bf16x16;
typedef __attribute__((ext_vector_type(8)))  float  f32x8;
typedef __attribute__((ext_vector_type(4)))  int    v4i;

// ---------------------------------------------------------------------------
// CDNA5 async global->LDS copy (ASYNCcnt path), with safe fallback.
// Builtin signature (from hipcc diagnostic): (v4i AS1*, v4i AS3*, Ii, Ii).
// ---------------------------------------------------------------------------
#ifndef USE_ASYNC_LDS
#define USE_ASYNC_LDS 1
#endif

#if USE_ASYNC_LDS && __has_builtin(__builtin_amdgcn_global_load_async_to_lds_b128) && \
    __has_builtin(__builtin_amdgcn_s_wait_asynccnt)
#define ASYNC_OK 1
typedef __attribute__((address_space(1))) void* as1_vp;
typedef __attribute__((address_space(3))) void* as3_vp;
typedef __attribute__((address_space(1))) v4i*  as1_v4i;
typedef __attribute__((address_space(3))) v4i*  as3_v4i;
#else
#define ASYNC_OK 0
#endif

// Stage a contiguous 4096-byte tile (16 rows x 128 bf16, ld==128) into LDS.
// 256 threads, one b128 chunk each. Includes the completion barrier.
__device__ __forceinline__ void stage_a_tile(const bf16* __restrict__ g, bf16* s) {
    int t = threadIdx.x;                   // 0..255
    const bf16* src = g + t * 8;           // 16 bytes per thread
    bf16*       dst = s + t * 8;
#if ASYNC_OK
    __builtin_amdgcn_global_load_async_to_lds_b128(
        (as1_v4i)(as1_vp)(void*)src, (as3_v4i)(as3_vp)(void*)dst, 0, 0);
    __builtin_amdgcn_s_wait_asynccnt(0);
#else
    *(uint4*)dst = *(const uint4*)src;
#endif
    __syncthreads();
}

union FragU { uint4 q[2]; bf16x16 v; };

// Load a 16-lane-wide WMMA fragment (A or B operand) from a row-major bf16
// matrix with leading dim `ld`. Per the CDNA5 ISA 16-bit A layout:
//   lanes 0-15  : row = r, elems 0..7 = K[k0..k0+7],   elems 8..15 = K[k0+16..k0+23]
//   lanes 16-31 : row = r, elems 0..7 = K[k0+8..k0+15], elems 8..15 = K[k0+24..k0+31]
__device__ __forceinline__ bf16x16 load_frag(const bf16* __restrict__ base,
                                             int row, int ld, int k0, int hi16) {
    const bf16* p = base + (size_t)row * ld + k0 + hi16 * 8;
    FragU u;
    u.q[0] = *(const uint4*)(p);
    u.q[1] = *(const uint4*)(p + 16);
    return u.v;
}

__device__ __forceinline__ f32x8 wmma_bf16(bf16x16 a, bf16x16 b, f32x8 c) {
    return __builtin_amdgcn_wmma_f32_16x16x32_bf16(false, a, false, b,
                                                   (short)0, c, false, false);
}

// ---------------------------------------------------------------------------
// Kernel: fp32 -> bf16 weight conversion (tiny, one-shot per launch)
// ---------------------------------------------------------------------------
__global__ void k_cvt_bf16(const float* __restrict__ in, bf16* __restrict__ out, int n) {
    int i = blockIdx.x * blockDim.x + threadIdx.x;
    if (i < n) out[i] = (bf16)in[i];
}

// ---------------------------------------------------------------------------
// Kernel: LayerNorm over C=128, emit bf16 activations. One wave32 per row.
// ---------------------------------------------------------------------------
__global__ void k_ln_bf16(const float* __restrict__ x,
                          const float* __restrict__ w,
                          const float* __restrict__ b,
                          bf16* __restrict__ out) {
    int row  = blockIdx.x * (blockDim.x >> 5) + (threadIdx.x >> 5);
    int lane = threadIdx.x & 31;
    const float* xr = x + (size_t)row * CDIM;
    float v[4];
#pragma unroll
    for (int i = 0; i < 4; ++i) v[i] = xr[lane + 32 * i];
    float s  = v[0] + v[1] + v[2] + v[3];
    float sq = v[0]*v[0] + v[1]*v[1] + v[2]*v[2] + v[3]*v[3];
#pragma unroll
    for (int off = 16; off > 0; off >>= 1) {
        s  += __shfl_xor(s,  off, 32);
        sq += __shfl_xor(sq, off, 32);
    }
    float mean = s * (1.0f / CDIM);
    float var  = sq * (1.0f / CDIM) - mean * mean;
    float rs   = rsqrtf(var + 1e-5f);
    bf16* orow = out + (size_t)row * CDIM;
#pragma unroll
    for (int i = 0; i < 4; ++i) {
        int c = lane + 32 * i;
        orow[c] = (bf16)((v[i] - mean) * rs * w[c] + b[c]);
    }
}

// ---------------------------------------------------------------------------
// Kernel: QKV projection.  qkv[M x 384] = xn[M x 128] @ qkv_w[384 x 128]^T + b.
// A tile (16x128) staged once in LDS per block; each wave owns 3 output tiles
// (cols wave*16 + {0,128,256}), so one A-fragment feeds 3 WMMAs per k-step.
// q-part (cols 0..127) scaled by d^-0.5.
// ---------------------------------------------------------------------------
__global__ void k_qkv(const bf16* __restrict__ xn,
                      const bf16* __restrict__ wq,
                      const float* __restrict__ bias,
                      bf16* __restrict__ qkv) {
    __shared__ bf16 aTile[16 * CDIM];
    int mBase = blockIdx.x * 16;
    stage_a_tile(xn + (size_t)mBase * CDIM, aTile);

    int wave  = threadIdx.x >> 5;
    int lane  = threadIdx.x & 31;
    int hi16  = lane >> 4;
    int r     = lane & 15;

    f32x8 acc[3] = {};
#pragma unroll
    for (int k0 = 0; k0 < CDIM; k0 += 32) {
        bf16x16 a = load_frag(aTile, r, CDIM, k0, hi16);
#pragma unroll
        for (int t = 0; t < 3; ++t) {
            int nBase = t * CDIM + wave * 16;
            bf16x16 w = load_frag(wq, nBase + r, CDIM, k0, hi16);
            acc[t] = wmma_bf16(a, w, acc[t]);
        }
    }
#pragma unroll
    for (int t = 0; t < 3; ++t) {
        float scale = (t == 0) ? 0.17677669529663687f : 1.0f;  // d^-0.5 on q only
        int col = t * CDIM + wave * 16 + r;
#pragma unroll
        for (int rr = 0; rr < 8; ++rr) {
            int grow = mBase + rr + 8 * hi16;
            qkv[(size_t)grow * 384 + col] = (bf16)((acc[t][rr] + bias[col]) * scale);
        }
    }
}

// ---------------------------------------------------------------------------
// Kernel: 1D neighborhood attention (window 7 over flattened N, clamped).
// One thread per (b, n, head).
// ---------------------------------------------------------------------------
__global__ void k_attn(const bf16* __restrict__ qkv,
                       const float* __restrict__ rpb,
                       bf16* __restrict__ out) {
    int t = blockIdx.x * blockDim.x + threadIdx.x;   // 0 .. MTOT*HEADS-1
    int h = t & (HEADS - 1);
    int n = (t >> 2) & (NSEQ - 1);
    int b = t >> 16;                                  // NSEQ*HEADS = 65536
    size_t row = (size_t)b * NSEQ + n;

    const bf16* qp = qkv + row * 384 + h * DHEAD;
    float q[DHEAD];
#pragma unroll
    for (int d = 0; d < DHEAD; ++d) q[d] = (float)qp[d];

    int start = n - (KS / 2);
    if (start < 0) start = 0;
    if (start > NSEQ - KS) start = NSEQ - KS;

    float logit[KS];
#pragma unroll
    for (int j = 0; j < KS; ++j) {
        const bf16* kp = qkv + ((size_t)b * NSEQ + start + j) * 384 + CDIM + h * DHEAD;
        float s = 0.f;
#pragma unroll
        for (int d = 0; d < DHEAD; ++d) s += q[d] * (float)kp[d];
        logit[j] = s + rpb[h * (2 * KS - 1) + (start + j - n + KS - 1)];
    }
    float mx = logit[0];
#pragma unroll
    for (int j = 1; j < KS; ++j) mx = fmaxf(mx, logit[j]);
    float denom = 0.f;
#pragma unroll
    for (int j = 0; j < KS; ++j) { logit[j] = __expf(logit[j] - mx); denom += logit[j]; }
    float inv = 1.0f / denom;

    float o[DHEAD];
#pragma unroll
    for (int d = 0; d < DHEAD; ++d) o[d] = 0.f;
#pragma unroll
    for (int j = 0; j < KS; ++j) {
        const bf16* vp = qkv + ((size_t)b * NSEQ + start + j) * 384 + 2 * CDIM + h * DHEAD;
        float p = logit[j] * inv;
#pragma unroll
        for (int d = 0; d < DHEAD; ++d) o[d] += p * (float)vp[d];
    }
    bf16* op = out + row * CDIM + h * DHEAD;
#pragma unroll
    for (int d = 0; d < DHEAD; ++d) op[d] = (bf16)o[d];
}

// ---------------------------------------------------------------------------
// Kernel: proj GEMM + residual + LayerNorm2, fused.
// A tile staged in LDS; block = 8 waves -> 16 rows x 128 cols (full C), so
// each row is complete in-block: LN2 runs on the LDS-resident fp32 tile.
// ---------------------------------------------------------------------------
__global__ void k_proj_res_ln2(const bf16* __restrict__ ao,
                               const bf16* __restrict__ wp,
                               const float* __restrict__ pb,
                               const float* __restrict__ x_in,
                               const float* __restrict__ n2w,
                               const float* __restrict__ n2b,
                               float* __restrict__ x2,
                               bf16* __restrict__ ln2) {
    __shared__ bf16  aTile[16 * CDIM];
    __shared__ float tile[16][CDIM];
    int mBase = blockIdx.x * 16;
    stage_a_tile(ao + (size_t)mBase * CDIM, aTile);

    int wave  = threadIdx.x >> 5;
    int lane  = threadIdx.x & 31;
    int hi16  = lane >> 4;
    int r     = lane & 15;
    int nBase = wave * 16;

    f32x8 acc = {};
#pragma unroll
    for (int k0 = 0; k0 < CDIM; k0 += 32) {
        bf16x16 a = load_frag(aTile, r, CDIM, k0, hi16);
        bf16x16 w = load_frag(wp, nBase + r, CDIM, k0, hi16);
        acc = wmma_bf16(a, w, acc);
    }
    int col = nBase + r;
#pragma unroll
    for (int rr = 0; rr < 8; ++rr) {
        int lrow = rr + 8 * hi16;
        int grow = mBase + lrow;
        float v  = acc[rr] + pb[col] + x_in[(size_t)grow * CDIM + col];
        tile[lrow][col] = v;
        x2[(size_t)grow * CDIM + col] = v;
    }
    __syncthreads();

    // LN2: 8 waves x 2 rows each
#pragma unroll
    for (int rr = 0; rr < 2; ++rr) {
        int lrow = wave * 2 + rr;
        float v[4];
#pragma unroll
        for (int i = 0; i < 4; ++i) v[i] = tile[lrow][lane + 32 * i];
        float s  = v[0] + v[1] + v[2] + v[3];
        float sq = v[0]*v[0] + v[1]*v[1] + v[2]*v[2] + v[3]*v[3];
#pragma unroll
        for (int off = 16; off > 0; off >>= 1) {
            s  += __shfl_xor(s,  off, 32);
            sq += __shfl_xor(sq, off, 32);
        }
        float mean = s * (1.0f / CDIM);
        float var  = sq * (1.0f / CDIM) - mean * mean;
        float rs   = rsqrtf(var + 1e-5f);
        bf16* orow = ln2 + (size_t)(mBase + lrow) * CDIM;
#pragma unroll
        for (int i = 0; i < 4; ++i) {
            int c = lane + 32 * i;
            orow[c] = (bf16)((v[i] - mean) * rs * n2w[c] + n2b[c]);
        }
    }
}

// ---------------------------------------------------------------------------
// Kernel: fused MLP.  fc1(16x512) -> GELU -> LDS -> fc2(16x128) + residual.
// A tile (4 KB) + hidden tile (16 KB) both LDS-resident (20 KB of 320 KB),
// eliminating a 268 MB HBM round trip for the 4C intermediate.
// ---------------------------------------------------------------------------
__global__ void k_mlp(const bf16* __restrict__ ln2,
                      const bf16* __restrict__ w1,
                      const float* __restrict__ b1,
                      const bf16* __restrict__ w2,
                      const float* __restrict__ b2,
                      const float* __restrict__ x2,
                      float* __restrict__ out) {
    __shared__ bf16 aTile[16 * CDIM];
    __shared__ bf16 h1[16 * 512];
    int mBase = blockIdx.x * 16;
    stage_a_tile(ln2 + (size_t)mBase * CDIM, aTile);

    int wave  = threadIdx.x >> 5;
    int lane  = threadIdx.x & 31;
    int hi16  = lane >> 4;
    int r     = lane & 15;

    // fc1 + GELU: each wave produces 4 tiles of 16x16 (block total 16 x 512);
    // one LDS A-fragment feeds 4 WMMAs per k-step.
    f32x8 acc1[4] = {};
#pragma unroll
    for (int k0 = 0; k0 < CDIM; k0 += 32) {
        bf16x16 a = load_frag(aTile, r, CDIM, k0, hi16);
#pragma unroll
        for (int t = 0; t < 4; ++t) {
            int nBase = wave * 64 + t * 16;
            bf16x16 w = load_frag(w1, nBase + r, CDIM, k0, hi16);
            acc1[t] = wmma_bf16(a, w, acc1[t]);
        }
    }
#pragma unroll
    for (int t = 0; t < 4; ++t) {
        int col = wave * 64 + t * 16 + r;
#pragma unroll
        for (int rr = 0; rr < 8; ++rr) {
            int lrow = rr + 8 * hi16;
            float v  = acc1[t][rr] + b1[col];
            v = 0.5f * v * (1.0f + erff(v * 0.70710678118654752f));  // exact GELU
            h1[lrow * 512 + col] = (bf16)v;
        }
    }
    __syncthreads();

    // fc2 + residual: one wave per 16x16 tile (8 tiles -> full 128 cols), K=512
    int nBase = wave * 16;
    f32x8 acc = {};
#pragma unroll
    for (int k0 = 0; k0 < 512; k0 += 32) {
        bf16x16 a = load_frag(h1, r, 512, k0, hi16);      // LDS-resident A
        bf16x16 w = load_frag(w2, nBase + r, 512, k0, hi16);
        acc = wmma_bf16(a, w, acc);
    }
    int col = nBase + r;
#pragma unroll
    for (int rr = 0; rr < 8; ++rr) {
        int grow = mBase + rr + 8 * hi16;
        out[(size_t)grow * CDIM + col] =
            acc[rr] + b2[col] + x2[(size_t)grow * CDIM + col];
    }
}

// ---------------------------------------------------------------------------
extern "C" void kernel_launch(void* const* d_in, const int* in_sizes, int n_in,
                              void* d_out, int out_size, void* d_ws, size_t ws_size,
                              hipStream_t stream) {
    const float* x       = (const float*)d_in[0];
    const float* n1w     = (const float*)d_in[1];
    const float* n1b     = (const float*)d_in[2];
    const float* qkv_w   = (const float*)d_in[3];
    const float* qkv_b   = (const float*)d_in[4];
    const float* rpb     = (const float*)d_in[5];
    const float* proj_w  = (const float*)d_in[6];
    const float* proj_b  = (const float*)d_in[7];
    const float* n2w     = (const float*)d_in[8];
    const float* n2b     = (const float*)d_in[9];
    const float* fc1_w   = (const float*)d_in[10];
    const float* fc1_b   = (const float*)d_in[11];
    const float* fc2_w   = (const float*)d_in[12];
    const float* fc2_b   = (const float*)d_in[13];

    char* ws = (char*)d_ws;
    size_t off = 0;
    bf16*  xn    = (bf16*)(ws + off);  off += (size_t)MTOT * CDIM * 2;      // 33.5 MB (reused as ln2)
    bf16*  qkv   = (bf16*)(ws + off);  off += (size_t)MTOT * 384  * 2;      // 100.7 MB
    bf16*  aout  = (bf16*)(ws + off);  off += (size_t)MTOT * CDIM * 2;      // 33.5 MB
    float* x2    = (float*)(ws + off); off += (size_t)MTOT * CDIM * 4;      // 67.1 MB
    bf16*  wqkvB = (bf16*)(ws + off);  off += (size_t)384 * 128 * 2;
    bf16*  wprojB= (bf16*)(ws + off);  off += (size_t)128 * 128 * 2;
    bf16*  wfc1B = (bf16*)(ws + off);  off += (size_t)512 * 128 * 2;
    bf16*  wfc2B = (bf16*)(ws + off);  off += (size_t)128 * 512 * 2;

    // Weight conversions (tiny)
    k_cvt_bf16<<<(384 * 128 + 255) / 256, 256, 0, stream>>>(qkv_w,  wqkvB,  384 * 128);
    k_cvt_bf16<<<(128 * 128 + 255) / 256, 256, 0, stream>>>(proj_w, wprojB, 128 * 128);
    k_cvt_bf16<<<(512 * 128 + 255) / 256, 256, 0, stream>>>(fc1_w,  wfc1B,  512 * 128);
    k_cvt_bf16<<<(128 * 512 + 255) / 256, 256, 0, stream>>>(fc2_w,  wfc2B,  128 * 512);

    // LN1 -> bf16 activations
    k_ln_bf16<<<MTOT / 8, 256, 0, stream>>>(x, n1w, n1b, xn);

    // QKV GEMM (WMMA bf16, LDS-staged A tile, async global->LDS)
    k_qkv<<<MTOT / 16, 256, 0, stream>>>(xn, wqkvB, qkv_b, qkv);

    // Neighborhood attention
    k_attn<<<(MTOT * HEADS) / 256, 256, 0, stream>>>(qkv, rpb, aout);

    // proj GEMM + residual + LN2 (fused)
    k_proj_res_ln2<<<MTOT / 16, 256, 0, stream>>>(aout, wprojB, proj_b, x,
                                                  n2w, n2b, x2, xn /* ln2 out */);

    // fused MLP (+ residual) -> final output
    k_mlp<<<MTOT / 16, 256, 0, stream>>>(xn, wfc1B, fc1_b, wfc2B, fc2_b, x2,
                                         (float*)d_out);
}
